// Decoder_33019708571890
// MI455X (gfx1250) — compile-verified, matrix operand order checked by reading, not run on hardware
//
#include <hip/hip_runtime.h>
#include <hip/hip_bf16.h>

#define DEV __device__ __forceinline__

typedef unsigned short u16t;
typedef __attribute__((ext_vector_type(16))) __bf16 v16bf;
typedef __attribute__((ext_vector_type(8)))  float  v8f;

struct U16x16 { u16t d[16]; };

// ---------- helpers ----------
DEV u16t f2bf(float f) {                       // fp32 -> bf16 round-to-nearest-even
  unsigned u = __builtin_bit_cast(unsigned, f);
  u += 0x7FFFu + ((u >> 16) & 1u);
  return (u16t)(u >> 16);
}

DEV void unpack8(uint4 g, u16t* d) {
  d[0] = (u16t)(g.x); d[1] = (u16t)(g.x >> 16);
  d[2] = (u16t)(g.y); d[3] = (u16t)(g.y >> 16);
  d[4] = (u16t)(g.z); d[5] = (u16t)(g.z >> 16);
  d[6] = (u16t)(g.w); d[7] = (u16t)(g.w >> 16);
}

// Load one 16-element bf16 fragment row (A- or B-layout): chunks at hi*8 and 16+hi*8.
// rowp must be 16B aligned; works for both LDS and global pointers.
DEV v16bf frag_ld(const u16t* rowp, int hi) {
  U16x16 t;
  uint4 a = *(const uint4*)(rowp + hi * 8);
  uint4 b = *(const uint4*)(rowp + 16 + hi * 8);
  unpack8(a, t.d);
  unpack8(b, t.d + 8);
  return __builtin_bit_cast(v16bf, t);
}

DEV v8f wmma_bf16(v16bf a, v16bf b, v8f c) {
  return __builtin_amdgcn_wmma_f32_16x16x32_bf16(false, a, false, b, (short)0, c, false, false);
}

DEV float gelu_exact(float x) {
  return 0.5f * x * (1.0f + erff(x * 0.70710678118654752f));
}

// ---------- GEMM: C = A(f32)[M,K] * B(f32)[K,N] (+epilogue), bf16 WMMA internally ----------
// Block tile 128x128, BK=64, double-buffered LDS, 8 waves x (32x64) wave tiles.
// MODE 0: out = bf16(acc + bias)
// MODE 1: out = f32 (acc + bias + res)
// MODE 2: out = f32 gelu(acc + bias)

#define GS_STRIDE 72              // u16 stride per row/col slice (144B: 16B-aligned, bank-spread)
#define GS_BUFSZ  (128 * GS_STRIDE)

DEV void stage_tile(const float* __restrict__ A, const float* __restrict__ Bw,
                    u16t* Asb, u16t* Bsb, int m0, int n0, int k0, int K, int N, int t) {
  // A tile: 128 rows x 64 k  (fp32 -> bf16)
#pragma unroll
  for (int i = 0; i < 8; ++i) {
    int f = i * 256 + t;
    int row = f >> 4, c4 = (f & 15) * 4;
    float4 v = *(const float4*)(A + (size_t)(m0 + row) * K + k0 + c4);
    ushort4 s;
    s.x = f2bf(v.x); s.y = f2bf(v.y); s.z = f2bf(v.z); s.w = f2bf(v.w);
    *(ushort4*)(Asb + row * GS_STRIDE + c4) = s;
  }
  // B tile transposed: 64 k x 128 n -> Bsb[n][k]
#pragma unroll
  for (int i = 0; i < 8; ++i) {
    int f = i * 256 + t;
    int kr = f >> 5, c4 = (f & 31) * 4;
    float4 v = *(const float4*)(Bw + (size_t)(k0 + kr) * N + n0 + c4);
    Bsb[(c4 + 0) * GS_STRIDE + kr] = f2bf(v.x);
    Bsb[(c4 + 1) * GS_STRIDE + kr] = f2bf(v.y);
    Bsb[(c4 + 2) * GS_STRIDE + kr] = f2bf(v.z);
    Bsb[(c4 + 3) * GS_STRIDE + kr] = f2bf(v.w);
  }
}

template <int MODE>
__global__ __launch_bounds__(256) void gemm_bf16_wmma(
    const float* __restrict__ A, const float* __restrict__ Bw,
    const float* __restrict__ bias, const float* __restrict__ res,
    void* __restrict__ outp, int M, int N, int K)
{
  (void)M;
  __shared__ u16t As[2 * GS_BUFSZ];
  __shared__ u16t Bs[2 * GS_BUFSZ];

  const int t  = threadIdx.x;
  const int m0 = blockIdx.y * 128;
  const int n0 = blockIdx.x * 128;
  const int w = t >> 5, lane = t & 31, lo = lane & 15, hi = lane >> 4;
  const int wm = (w & 3) * 32;    // 4x2 wave grid of 32x64 tiles
  const int wn = (w >> 2) * 64;

  const v8f vzero = {0.f, 0.f, 0.f, 0.f, 0.f, 0.f, 0.f, 0.f};
  v8f acc[2][4];
#pragma unroll
  for (int mt = 0; mt < 2; ++mt)
#pragma unroll
    for (int nt = 0; nt < 4; ++nt) acc[mt][nt] = vzero;

  stage_tile(A, Bw, As, Bs, m0, n0, 0, K, N, t);

  int buf = 0;
  for (int k0 = 0; k0 < K; k0 += 64) {
    __syncthreads();
    const u16t* Asb = As + buf * GS_BUFSZ;
    const u16t* Bsb = Bs + buf * GS_BUFSZ;

    if (k0 + 64 < K) {
      stage_tile(A, Bw, As + (buf ^ 1) * GS_BUFSZ, Bs + (buf ^ 1) * GS_BUFSZ,
                 m0, n0, k0 + 64, K, N, t);
      if (k0 + 128 < K) {
        // pull the tile after next into L2 (global_prefetch_b8)
        __builtin_prefetch(A + (size_t)(m0 + (t >> 1)) * K + (k0 + 128) + (t & 1) * 32, 0, 1);
        __builtin_prefetch(Bw + (size_t)(k0 + 128 + (t >> 2)) * N + n0 + (t & 3) * 32, 0, 1);
      }
    }

#pragma unroll
    for (int s = 0; s < 2; ++s) {
      v16bf af0 = frag_ld(Asb + (wm + lo) * GS_STRIDE + s * 32, hi);
      v16bf af1 = frag_ld(Asb + (wm + 16 + lo) * GS_STRIDE + s * 32, hi);
#pragma unroll
      for (int nt = 0; nt < 4; ++nt) {
        v16bf bfr = frag_ld(Bsb + (wn + nt * 16 + lo) * GS_STRIDE + s * 32, hi);
        acc[0][nt] = wmma_bf16(af0, bfr, acc[0][nt]);
        acc[1][nt] = wmma_bf16(af1, bfr, acc[1][nt]);
      }
    }
    buf ^= 1;
  }

  // epilogue (C layout: row = r + hi*8, col = lo within each 16x16 tile)
#pragma unroll
  for (int nt = 0; nt < 4; ++nt) {
    int gcol = n0 + wn + nt * 16 + lo;
    float bv = bias[gcol];
#pragma unroll
    for (int mt = 0; mt < 2; ++mt) {
#pragma unroll
      for (int r = 0; r < 8; ++r) {
        int grow = m0 + wm + mt * 16 + r + hi * 8;
        float v = acc[mt][nt][r] + bv;
        if (MODE == 0) {
          ((u16t*)outp)[(size_t)grow * N + gcol] = f2bf(v);
        } else if (MODE == 1) {
          v += res[(size_t)grow * N + gcol];
          ((float*)outp)[(size_t)grow * N + gcol] = v;
        } else {
          ((float*)outp)[(size_t)grow * N + gcol] = gelu_exact(v);
        }
      }
    }
  }
}

// ---------- Flash-style attention: one wave = 16 q rows of one (b, head) ----------
__global__ __launch_bounds__(256) void attn_kernel(
    const u16t* __restrict__ qb, const u16t* __restrict__ kb,
    const u16t* __restrict__ vb, const int* __restrict__ mask,
    float* __restrict__ obuf)
{
  const int S = 1024, D = 1024;
  __shared__ u16t smem[8 * 2560];   // per wave: VT (64dh x 32kv) + P (16q x 32kv)

  const int t = threadIdx.x, w = t >> 5, lane = t & 31, lo = lane & 15, hi = lane >> 4;
  const int task = blockIdx.x * 8 + w;      // B*H*(S/16) = 2048 tasks
  const int b   = task >> 10;               // 1024 tasks per batch
  const int rem = task & 1023;
  const int hh  = rem >> 6;                 // head
  const int q0  = (rem & 63) << 4;          // q block of 16

  u16t* VT = smem + w * 2560;               // [dh 0..63][kv 0..31]
  u16t* Pl = VT + 2048;                     // [q 0..15][kv 0..31]

  const size_t bS = (size_t)b * S;
  const int hcol = hh * 64;
  const v8f vzero = {0.f, 0.f, 0.f, 0.f, 0.f, 0.f, 0.f, 0.f};

  // preload Q fragments (A-layout, dh slices 0-31 and 32-63)
  v16bf qa[2];
#pragma unroll
  for (int ds = 0; ds < 2; ++ds)
    qa[ds] = frag_ld(qb + (bS + q0 + lo) * D + hcol + ds * 32, hi);

  v8f o[4];
#pragma unroll
  for (int dt = 0; dt < 4; ++dt) o[dt] = vzero;
  float m[8], l[8];
#pragma unroll
  for (int r = 0; r < 8; ++r) { m[r] = -1e30f; l[r] = 0.f; }

  const int* mrow = mask + (size_t)b * S * S;

  for (int kv0 = 0; kv0 < S; kv0 += 32) {
    // stage V block transposed into LDS: lane owns kv row kv0+lane
    const u16t* vp = vb + (bS + kv0 + lane) * D + hcol;
#pragma unroll
    for (int ch = 0; ch < 8; ++ch) {
      u16t d[8];
      unpack8(*(const uint4*)(vp + ch * 8), d);
#pragma unroll
      for (int j = 0; j < 8; ++j) VT[(ch * 8 + j) * 32 + lane] = d[j];
    }
    __syncthreads();

    // scores: S16x32 = Q(16x64) * K(32x64)^T  (B-frag rows of K are contiguous)
    v8f sf[2];
#pragma unroll
    for (int sub = 0; sub < 2; ++sub) {
      sf[sub] = vzero;
#pragma unroll
      for (int ds = 0; ds < 2; ++ds) {
        v16bf kf = frag_ld(kb + (bS + kv0 + sub * 16 + lo) * D + hcol + ds * 32, hi);
        sf[sub] = wmma_bf16(qa[ds], kf, sf[sub]);
      }
    }

    // scale + mask
    float s[2][8], corr[8];
#pragma unroll
    for (int sub = 0; sub < 2; ++sub)
#pragma unroll
      for (int r = 0; r < 8; ++r) {
        float x = sf[sub][r] * 0.125f;   // 1/sqrt(64)
        int mq = q0 + r + hi * 8;
        if (mrow[(size_t)mq * S + kv0 + sub * 16 + lo] == 0) x = -1e30f;
        s[sub][r] = x;
      }

    // online softmax (row stats replicated across each 16-lane half)
#pragma unroll
    for (int r = 0; r < 8; ++r) {
      float rm = fmaxf(s[0][r], s[1][r]);
#pragma unroll
      for (int d = 1; d < 16; d <<= 1) rm = fmaxf(rm, __shfl_xor(rm, d, 32));
      float nm = fmaxf(m[r], rm);
      corr[r] = __expf(m[r] - nm);
      m[r] = nm;
      float p0 = __expf(s[0][r] - nm);
      float p1 = __expf(s[1][r] - nm);
      s[0][r] = p0; s[1][r] = p1;
      float rs = p0 + p1;
#pragma unroll
      for (int d = 1; d < 16; d <<= 1) rs += __shfl_xor(rs, d, 32);
      l[r] = l[r] * corr[r] + rs;
    }
#pragma unroll
    for (int dt = 0; dt < 4; ++dt)
#pragma unroll
      for (int r = 0; r < 8; ++r) o[dt][r] *= corr[r];

    // P (C-layout) -> LDS -> A-layout fragment
#pragma unroll
    for (int sub = 0; sub < 2; ++sub)
#pragma unroll
      for (int r = 0; r < 8; ++r)
        Pl[(r + hi * 8) * 32 + sub * 16 + lo] = f2bf(s[sub][r]);
    __syncthreads();

    v16bf pf = frag_ld(Pl + lo * 32, hi);
#pragma unroll
    for (int dt = 0; dt < 4; ++dt) {
      v16bf vf = frag_ld(VT + (dt * 16 + lo) * 32, hi);
      o[dt] = wmma_bf16(pf, vf, o[dt]);
    }
  }

#pragma unroll
  for (int r = 0; r < 8; ++r) l[r] = 1.0f / l[r];
#pragma unroll
  for (int dt = 0; dt < 4; ++dt)
#pragma unroll
    for (int r = 0; r < 8; ++r)
      obuf[(bS + q0 + r + hi * 8) * D + hcol + dt * 16 + lo] = o[dt][r] * l[r];
}

// ---------- LayerNorm over last dim (D=1024), one block per row ----------
__global__ __launch_bounds__(256) void ln_kernel(
    const float* __restrict__ in, const float* __restrict__ g,
    const float* __restrict__ bt, float* __restrict__ out)
{
  const int D = 1024;
  const int row = blockIdx.x, t = threadIdx.x;
  __shared__ float red[256];
  const float* ip = in + (size_t)row * D;

  float v[4];
#pragma unroll
  for (int i = 0; i < 4; ++i) v[i] = ip[t + i * 256];
  float s = v[0] + v[1] + v[2] + v[3];
  red[t] = s; __syncthreads();
  for (int off = 128; off > 0; off >>= 1) {
    if (t < off) red[t] += red[t + off];
    __syncthreads();
  }
  float mean = red[0] * (1.0f / D);
  __syncthreads();

  float q = 0.f;
#pragma unroll
  for (int i = 0; i < 4; ++i) { float d = v[i] - mean; q += d * d; }
  red[t] = q; __syncthreads();
  for (int off = 128; off > 0; off >>= 1) {
    if (t < off) red[t] += red[t + off];
    __syncthreads();
  }
  float rs = rsqrtf(red[0] * (1.0f / D) + 1e-5f);

  float* op = out + (size_t)row * D;
#pragma unroll
  for (int i = 0; i < 4; ++i) {
    int c = t + i * 256;
    op[c] = (v[i] - mean) * rs * g[c] + bt[c];
  }
}

// ---------- launcher ----------
extern "C" void kernel_launch(void* const* d_in, const int* in_sizes, int n_in,
                              void* d_out, int out_size, void* d_ws, size_t ws_size,
                              hipStream_t stream) {
  (void)in_sizes; (void)n_in; (void)out_size; (void)ws_size;
  const int L = 6, Bn = 2, S = 1024, D = 1024, F = 4096;
  const int M = Bn * S;  // 2048

  const float* x    = (const float*)d_in[0];
  const int*   mask = (const int*)  d_in[1];
  const float* Wq = (const float*)d_in[2];  const float* bq = (const float*)d_in[3];
  const float* Wk = (const float*)d_in[4];  const float* bk = (const float*)d_in[5];
  const float* Wv = (const float*)d_in[6];  const float* bv = (const float*)d_in[7];
  const float* Wo = (const float*)d_in[8];  const float* bo = (const float*)d_in[9];
  const float* g1 = (const float*)d_in[10]; const float* b1n = (const float*)d_in[11];
  const float* W1 = (const float*)d_in[12]; const float* b1m = (const float*)d_in[13];
  const float* W2 = (const float*)d_in[14]; const float* b2m = (const float*)d_in[15];
  const float* g2 = (const float*)d_in[16]; const float* b2n = (const float*)d_in[17];

  char* ws = (char*)d_ws;
  float* h  = (float*)(ws);                          // 8 MB
  float* t0 = (float*)(ws + ((size_t)8  << 20));     // 8 MB
  u16t*  qb = (u16t*) (ws + ((size_t)16 << 20));     // 4 MB
  u16t*  kb2= (u16t*) (ws + ((size_t)20 << 20));     // 4 MB
  u16t*  vb2= (u16t*) (ws + ((size_t)24 << 20));     // 4 MB
  float* ob = (float*)(ws + ((size_t)28 << 20));     // 8 MB
  float* t2 = (float*)(ws + ((size_t)36 << 20));     // 32 MB

  hipMemcpyAsync(h, x, (size_t)M * D * sizeof(float), hipMemcpyDeviceToDevice, stream);

  dim3 blk(256);
  dim3 gD(D / 128, M / 128);   // N = D
  dim3 gF(F / 128, M / 128);   // N = F
  dim3 gAttn(256);             // 2048 wave-tasks / 8 waves per block
  dim3 gLN(M);

  for (int l = 0; l < L; ++l) {
    const float* wq = Wq + (size_t)l * D * D; const float* bqv = bq + (size_t)l * D;
    const float* wk = Wk + (size_t)l * D * D; const float* bkv = bk + (size_t)l * D;
    const float* wv = Wv + (size_t)l * D * D; const float* bvv = bv + (size_t)l * D;
    const float* wo = Wo + (size_t)l * D * D; const float* bov = bo + (size_t)l * D;
    const float* w1 = W1 + (size_t)l * D * F; const float* b1v = b1m + (size_t)l * F;
    const float* w2 = W2 + (size_t)l * F * D; const float* b2v = b2m + (size_t)l * D;
    const float* g1v = g1 + (size_t)l * D; const float* b1l = b1n + (size_t)l * D;
    const float* g2v = g2 + (size_t)l * D; const float* b2l = b2n + (size_t)l * D;

    gemm_bf16_wmma<0><<<gD, blk, 0, stream>>>(h, wq, bqv, nullptr, qb,  M, D, D);
    gemm_bf16_wmma<0><<<gD, blk, 0, stream>>>(h, wk, bkv, nullptr, kb2, M, D, D);
    gemm_bf16_wmma<0><<<gD, blk, 0, stream>>>(h, wv, bvv, nullptr, vb2, M, D, D);

    attn_kernel<<<gAttn, blk, 0, stream>>>(qb, kb2, vb2, mask, ob);

    gemm_bf16_wmma<1><<<gD, blk, 0, stream>>>(ob, wo, bov, h, t0, M, D, D);
    ln_kernel<<<gLN, blk, 0, stream>>>(t0, g1v, b1l, h);

    gemm_bf16_wmma<2><<<gF, blk, 0, stream>>>(h, w1, b1v, nullptr, t2, M, F, D);
    gemm_bf16_wmma<1><<<gD, blk, 0, stream>>>(t2, w2, b2v, h, t0, M, D, F);
    ln_kernel<<<gLN, blk, 0, stream>>>(t0, g2v, b2l, h);
  }

  hipMemcpyAsync(d_out, h, (size_t)M * D * sizeof(float), hipMemcpyDeviceToDevice, stream);
}
